// MAB_57475252355499
// MI455X (gfx1250) — matrix operand order repaired and checked by reading, not compiled, not run on hardware
//
#include <hip/hip_runtime.h>
#include <math.h>

#define B_   32
#define NQ_  1024
#define NK_  1024
#define DV_  512
#define H_   8
#define DH_  64
#define NTOK_ (B_ * NQ_)   // 32768 GEMM rows, K = N = DV_ for every GEMM

typedef __attribute__((ext_vector_type(16))) __bf16 v16bf;
typedef __attribute__((ext_vector_type(8)))  __bf16 v8bf;
typedef __attribute__((ext_vector_type(2)))  __bf16 v2bf;
typedef __attribute__((ext_vector_type(8)))  float  v8f;

static __device__ __forceinline__ v8f wmma_bf16(v16bf a, v16bf b, v8f c) {
  // D = A(16x32 bf16) x B(32x16 bf16) + C(16x16 f32)
  return __builtin_amdgcn_wmma_f32_16x16x32_bf16(false, a, false, b, (short)0, c,
                                                 false, false);
}
// contiguous 32B fragment load (caller guarantees 32B alignment)
static __device__ __forceinline__ v16bf ld16(const __bf16* p) {
  return *(const v16bf*)p;
}
// fragment from two contiguous 16B halves (16B aligned)
static __device__ __forceinline__ v16bf ld8x2(const __bf16* p0, const __bf16* p1) {
  v8bf lo = *(const v8bf*)p0;
  v8bf hi = *(const v8bf*)p1;
  return __builtin_shufflevector(lo, hi, 0, 1, 2, 3, 4, 5, 6, 7,
                                 8, 9, 10, 11, 12, 13, 14, 15);
}

// ===========================================================================
// GEMM: C = A[32768,512](f32) @ W[512,512](f32) + bias. bf16 WMMA, fp32 acc.
// Tiles 128x128x32, 256 threads = 8 waves, wave -> 32x64 of the output tile.
// A slab row-major (stride 40), W slab TRANSPOSED in LDS (stride 40) so both
// fragment types are contiguous b128 DS loads.
// MODE 0: Cf fp32 [M,N];  MODE 1: Cb bf16 [M,N];  MODE 2: Cb bf16 transposed
// per-batch [B][N][1024] (for v^T).  Compile-time MODE: no epilogue branches.
// ===========================================================================
#define AS_STRIDE 40
#define BT_STRIDE 40
template <int MODE>
__global__ __launch_bounds__(256) void gemm_bias_kernel(
    const float* __restrict__ A, const float* __restrict__ W,
    const float* __restrict__ bias, float* __restrict__ Cf,
    __bf16* __restrict__ Cb) {
  constexpr int N = DV_, K = DV_;
  __shared__ __bf16 As[128 * AS_STRIDE];
  __shared__ __bf16 Bt[128 * BT_STRIDE];
  const int t = threadIdx.x, lane = t & 31, w = t >> 5;
  const int m0 = blockIdx.y * 128, n0 = blockIdx.x * 128;
  const int wm = (w >> 1) * 32;  // 0,32,64,96
  const int wn = (w & 1) * 64;   // 0,64
  v8f acc[2][4] = {};

  for (int k0 = 0; k0 < K; k0 += 32) {
    __syncthreads();
    if (k0 + 32 < K) {  // global_prefetch_b8 of next K-slab
      __builtin_prefetch(&A[(size_t)(m0 + (t >> 1)) * K + k0 + 32], 0, 0);
      __builtin_prefetch(&W[(size_t)(k0 + 32 + (t >> 3)) * N + n0], 0, 0);
    }
    // A slab 128x32: packed pair converts (fp32x2 -> v2bf), 8 pairs/thread
#pragma unroll
    for (int i = 0; i < 8; ++i) {
      const int idx = t + i * 256, r = idx >> 4, c = (idx & 15) * 2;
      const float* ap = &A[(size_t)(m0 + r) * K + (k0 + c)];
      v2bf p;
      p[0] = (__bf16)ap[0];
      p[1] = (__bf16)ap[1];
      *(v2bf*)&As[r * AS_STRIDE + c] = p;
    }
    // W slab 32x128, stored transposed: coalesced global read, scattered DS
#pragma unroll
    for (int i = 0; i < 16; ++i) {
      const int idx = t + i * 256, kk = idx >> 7, n = idx & 127;
      Bt[n * BT_STRIDE + kk] = (__bf16)W[(size_t)(k0 + kk) * N + (n0 + n)];
    }
    __syncthreads();
    v16bf af[2], bfr[4];
#pragma unroll
    for (int i = 0; i < 2; ++i) {
      const int m = wm + i * 16 + (lane & 15), s0 = (lane >> 4) * 8;
      af[i] = ld8x2(&As[m * AS_STRIDE + s0], &As[m * AS_STRIDE + 16 + s0]);
    }
#pragma unroll
    for (int j = 0; j < 4; ++j) {
      const int n = wn + j * 16 + (lane & 15), ks = (lane >> 4) * 8;
      bfr[j] = ld8x2(&Bt[n * BT_STRIDE + ks], &Bt[n * BT_STRIDE + 16 + ks]);
    }
#pragma unroll
    for (int i = 0; i < 2; ++i)
#pragma unroll
      for (int j = 0; j < 4; ++j) acc[i][j] = wmma_bf16(af[i], bfr[j], acc[i][j]);
  }
  // epilogue: lane holds col n; VGPR r holds row r + 8*(lane>>4)
#pragma unroll
  for (int i = 0; i < 2; ++i)
#pragma unroll
    for (int j = 0; j < 4; ++j) {
      const int col = n0 + wn + j * 16 + (lane & 15);
      const float bv = bias[col];
#pragma unroll
      for (int r = 0; r < 8; ++r) {
        const int row = m0 + wm + i * 16 + r + 8 * (lane >> 4);
        const float val = acc[i][j][r] + bv;
        if constexpr (MODE == 0) {
          Cf[(size_t)row * N + col] = val;
        } else if constexpr (MODE == 1) {
          Cb[(size_t)row * N + col] = (__bf16)val;
        } else {  // transposed per batch: [b][col][row%1024]
          Cb[((size_t)(row >> 10) * N + col) * NK_ + (row & 1023)] = (__bf16)val;
        }
      }
    }
}

// ===========================================================================
// Pass 1: column-softmax stats over the query axis. Wave owns 16 k-columns;
// lane pair (l, l^16) shares a column -> shfl_xor(16) reduction.
// All fragments are contiguous vector loads of bf16 q/k.
// ===========================================================================
__global__ __launch_bounds__(256) void attn_colstats_kernel(
    const __bf16* __restrict__ q, const __bf16* __restrict__ k,
    float* __restrict__ mstat, float* __restrict__ lstat) {
  const int t = threadIdx.x, lane = t & 31, w = t >> 5;
  const int h = blockIdx.z, b = blockIdx.y;
  const int kw = blockIdx.x * 128 + w * 16;
  const __bf16* qb = q + (size_t)b * NQ_ * DV_ + h * DH_;
  const __bf16* kb = k + (size_t)b * NK_ * DV_ + h * DH_;
  const float scale = 0.044194173824159216f;  // 1/sqrt(512)

  // B = kh^T fragments: lane = column kcol, contiguous 16-d segment
  const int ds = (lane >> 4) * 16;
  const __bf16* kr = kb + (size_t)(kw + (lane & 15)) * DV_;
  const v16bf bf0 = ld16(kr + ds);
  const v16bf bf1 = ld16(kr + 32 + ds);

  float mrun = -INFINITY, lrun = 0.f;
  for (int q0 = 0; q0 < NQ_; q0 += 16) {
    const __bf16* qr = qb + (size_t)(q0 + (lane & 15)) * DV_;
    const int s0 = (lane >> 4) * 16;
    const v16bf a0 = ld8x2(qr + s0, qr + 8 + s0);
    const v16bf a1 = ld8x2(qr + 32 + s0, qr + 40 + s0);
    v8f s = {};
    s = wmma_bf16(a0, bf0, s);
    s = wmma_bf16(a1, bf1, s);
    float sv[8];
#pragma unroll
    for (int r = 0; r < 8; ++r) sv[r] = s[r] * scale;
    float tm = sv[0];
#pragma unroll
    for (int r = 1; r < 8; ++r) tm = fmaxf(tm, sv[r]);
    tm = fmaxf(tm, __shfl_xor(tm, 16));  // partner lane: same column
    const float nm = fmaxf(mrun, tm);
    float ps = 0.f;
#pragma unroll
    for (int r = 0; r < 8; ++r) ps += __expf(sv[r] - nm);
    ps += __shfl_xor(ps, 16);
    lrun = lrun * __expf(mrun - nm) + ps;
    mrun = nm;
  }
  if (lane < 16) {
    const size_t sidx = ((size_t)h * B_ + b) * NK_ + (kw + lane);
    mstat[sidx] = mrun;
    lstat[sidx] = lrun;
  }
}

// ===========================================================================
// Pass 2: O = qh + softmax_cols(S) @ vh.  Wave owns a 16-q tile; per 32-k
// chunk: WMMA scores -> exp/normalize -> per-wave LDS transpose (C-layout ->
// A-layout; same-wave DS ops are in-order + explicit s_wait_dscnt) -> WMMA
// against v^T (contiguous B-fragments thanks to transposed v storage).
// ===========================================================================
__global__ __launch_bounds__(256) void attn_out_kernel(
    const __bf16* __restrict__ q, const __bf16* __restrict__ k,
    const __bf16* __restrict__ vT, const float* __restrict__ mstat,
    const float* __restrict__ lstat, float* __restrict__ outp) {
  __shared__ __bf16 Achunk[8][16 * 32];  // per-wave scratch
  const int t = threadIdx.x, lane = t & 31, w = t >> 5;
  const int h = blockIdx.z, b = blockIdx.y;
  const int q0 = blockIdx.x * 128 + w * 16;
  const __bf16* qb = q + (size_t)b * NQ_ * DV_ + h * DH_;
  const __bf16* kb = k + (size_t)b * NK_ * DV_ + h * DH_;
  const __bf16* vt = vT + ((size_t)b * DV_ + h * DH_) * NK_;
  const size_t sbase = ((size_t)h * B_ + b) * NK_;
  const float scale = 0.044194173824159216f;
  __bf16* ach = &Achunk[w][0];

  // loop-invariant qh A-fragments (16 q rows x 64 d)
  const int s0 = (lane >> 4) * 16;
  const __bf16* qr = qb + (size_t)(q0 + (lane & 15)) * DV_;
  const v16bf qa0 = ld8x2(qr + s0, qr + 8 + s0);
  const v16bf qa1 = ld8x2(qr + 32 + s0, qr + 40 + s0);
  v8f acc[4] = {};

  for (int k0 = 0; k0 < NK_; k0 += 32) {
#pragma unroll
    for (int sub = 0; sub < 2; ++sub) {
      const int kcol = k0 + sub * 16 + (lane & 15);
      const __bf16* kr = kb + (size_t)kcol * DV_;
      const int ds = (lane >> 4) * 16;
      const v16bf b0 = ld16(kr + ds);
      const v16bf b1 = ld16(kr + 32 + ds);
      v8f s = {};
      s = wmma_bf16(qa0, b0, s);
      s = wmma_bf16(qa1, b1, s);
      const float mval  = mstat[sbase + kcol];
      const float inv_l = 1.f / lstat[sbase + kcol];
      const int colc = sub * 16 + (lane & 15);
#pragma unroll
      for (int r = 0; r < 8; ++r) {
        const int rowc = r + 8 * (lane >> 4);
        ach[rowc * 32 + colc] = (__bf16)(__expf(s[r] * scale - mval) * inv_l);
      }
    }
    asm volatile("s_wait_dscnt 0" ::: "memory");  // LDS transpose round-trip
    const int m = lane & 15;
    const v16bf af = ld8x2(&ach[m * 32 + s0], &ach[m * 32 + 8 + s0]);
#pragma unroll
    for (int dt = 0; dt < 4; ++dt) {
      const int dcol = dt * 16 + (lane & 15);
      const int ks = (lane >> 4) * 16;
      const v16bf bv = ld16(vt + (size_t)dcol * NK_ + k0 + ks);
      acc[dt] = wmma_bf16(af, bv, acc[dt]);
    }
  }
  // residual + merged-head store
#pragma unroll
  for (int dt = 0; dt < 4; ++dt) {
    const int dcol = dt * 16 + (lane & 15);
#pragma unroll
    for (int r = 0; r < 8; ++r) {
      const int row = q0 + r + 8 * (lane >> 4);
      outp[((size_t)b * NQ_ + row) * DV_ + h * DH_ + dcol] =
          acc[dt][r] + (float)qb[(size_t)row * DV_ + dcol];
    }
  }
}

// ===========================================================================
// LayerNorm (optionally fused with z = x + relu(y)). 256 thr, row of 512.
// ===========================================================================
__global__ __launch_bounds__(256) void ln_kernel(
    const float* __restrict__ x, const float* __restrict__ y,
    const float* __restrict__ g, const float* __restrict__ be,
    float* __restrict__ outp, int fuse) {
  __shared__ float redS[8], redQ[8];
  const int row = blockIdx.x, t = threadIdx.x;
  const float* xr = x + (size_t)row * DV_;
  float v0 = xr[t], v1 = xr[t + 256];
  if (fuse) {
    const float* yr = y + (size_t)row * DV_;
    v0 += fmaxf(yr[t], 0.f);
    v1 += fmaxf(yr[t + 256], 0.f);
  }
  float s = v0 + v1, ss = v0 * v0 + v1 * v1;
#pragma unroll
  for (int o = 16; o > 0; o >>= 1) { s += __shfl_xor(s, o); ss += __shfl_xor(ss, o); }
  const int w = t >> 5, lane = t & 31;
  if (lane == 0) { redS[w] = s; redQ[w] = ss; }
  __syncthreads();
  float ts = 0.f, tss = 0.f;
#pragma unroll
  for (int i = 0; i < 8; ++i) { ts += redS[i]; tss += redQ[i]; }
  const float mu  = ts * (1.f / DV_);
  const float var = tss * (1.f / DV_) - mu * mu;
  const float inv = rsqrtf(var + 1e-5f);
  outp[(size_t)row * DV_ + t]       = (v0 - mu) * inv * g[t] + be[t];
  outp[(size_t)row * DV_ + t + 256] = (v1 - mu) * inv * g[t + 256] + be[t + 256];
}

// ===========================================================================
extern "C" void kernel_launch(void* const* d_in, const int* in_sizes, int n_in,
                              void* d_out, int out_size, void* d_ws, size_t ws_size,
                              hipStream_t stream) {
  (void)in_sizes; (void)n_in; (void)out_size; (void)ws_size;
  const float* Q   = (const float*)d_in[0];
  const float* K   = (const float*)d_in[1];
  const float* Wq  = (const float*)d_in[2];
  const float* bq  = (const float*)d_in[3];
  const float* Wk  = (const float*)d_in[4];
  const float* bk  = (const float*)d_in[5];
  const float* Wv  = (const float*)d_in[6];
  const float* bv  = (const float*)d_in[7];
  const float* Wo  = (const float*)d_in[8];
  const float* bo  = (const float*)d_in[9];
  const float* g0  = (const float*)d_in[10];
  const float* be0 = (const float*)d_in[11];
  const float* g1  = (const float*)d_in[12];
  const float* be1 = (const float*)d_in[13];

  const size_t BUF = (size_t)NTOK_ * DV_;  // 16Mi elements per buffer
  __bf16* qb    = (__bf16*)d_ws;           // bf16, 32 MB
  __bf16* kbb   = qb + BUF;                // bf16, 32 MB
  __bf16* vTb   = kbb + BUF;               // bf16 transposed [B][DV][NK], 32 MB
  float*  attn  = (float*)(vTb + BUF);     // fp32, 64 MB
  float*  mstat = attn + BUF;              // H*B*NK
  float*  lstat = mstat + (size_t)H_ * B_ * NK_;
  float*  x0    = (float*)d_ws;            // overlays dead qb+kbb (64 MB)
  float*  ybuf  = attn;                    // overlays dead attn

  const dim3 gemm_grid(DV_ / 128, NTOK_ / 128);  // 4 x 256
  gemm_bias_kernel<1><<<gemm_grid, 256, 0, stream>>>(Q, Wq, bq, nullptr, qb);
  gemm_bias_kernel<1><<<gemm_grid, 256, 0, stream>>>(K, Wk, bk, nullptr, kbb);
  gemm_bias_kernel<2><<<gemm_grid, 256, 0, stream>>>(K, Wv, bv, nullptr, vTb);

  attn_colstats_kernel<<<dim3(NK_ / 128, B_, H_), 256, 0, stream>>>(qb, kbb,
                                                                    mstat, lstat);
  attn_out_kernel<<<dim3(NQ_ / 128, B_, H_), 256, 0, stream>>>(qb, kbb, vTb,
                                                               mstat, lstat, attn);

  ln_kernel<<<(unsigned)NTOK_, 256, 0, stream>>>(attn, nullptr, g0, be0, x0, 0);
  gemm_bias_kernel<0><<<gemm_grid, 256, 0, stream>>>(x0, Wo, bo, ybuf, nullptr);
  ln_kernel<<<(unsigned)NTOK_, 256, 0, stream>>>(x0, ybuf, g1, be1, (float*)d_out, 1);
}
